// Attention_80444737454255
// MI455X (gfx1250) — matrix operand order repaired and checked by reading, not compiled, not run on hardware
//
#include <hip/hip_runtime.h>
#include <hip/hip_bf16.h>

typedef __attribute__((ext_vector_type(16))) _Float16 v16h;
typedef __attribute__((ext_vector_type(8)))  _Float16 v8h;
typedef __attribute__((ext_vector_type(8)))  float    v8f;
typedef __attribute__((ext_vector_type(4)))  float    v4f;

#define DIMX   768
#define NHEAD  12
#define HDIM   64
#define BATCH  32
#define SEQ    577
#define ATTN_SCALE 0.125f   // 64^-0.5

__device__ __forceinline__ v8f wmma16x16x32(v16h a, v16h b, v8f c) {
  return __builtin_amdgcn_wmma_f32_16x16x32_f16(false, a, false, b, (short)0, c,
                                                false, false);
}

__device__ __forceinline__ v16h cat8(v8h lo, v8h hi) {
  v16h r;
#pragma unroll
  for (int i = 0; i < 8; ++i) { r[i] = lo[i]; r[i + 8] = hi[i]; }
  return r;
}

// CDNA5 async global->LDS copy (GVS): writes LDS at vdst + offset,
// reads global at saddr + vaddr + offset. Tracked on ASYNCcnt.
#define ASYNC_LD(lds, voff, sbase, OFF)                                  \
  asm volatile("global_load_async_to_lds_b128 %0, %1, %2 offset:" OFF    \
               :: "v"(lds), "v"(voff), "s"(sbase) : "memory")

__device__ __forceinline__ void wait_async0() {
  asm volatile("s_wait_asynccnt 0x0" ::: "memory");
}

// ---------------------------------------------------------------------------
// f32 -> f16 elementwise conversion (one-shot pass so all GEMM staging can be
// pure async copies). n8 = elements / 8.
// ---------------------------------------------------------------------------
__global__ __launch_bounds__(256) void cvt_f32_f16(
    const float* __restrict__ in, _Float16* __restrict__ out, int n8) {
  const int i = blockIdx.x * blockDim.x + threadIdx.x;
  if (i < n8) {
    const v4f* p = (const v4f*)(in + (size_t)i * 8);
    const v4f a = p[0], b = p[1];
    v8h o;
#pragma unroll
    for (int j = 0; j < 4; ++j) {
      o[j] = (_Float16)a[j];
      o[j + 4] = (_Float16)b[j];
    }
    *(v8h*)(out + (size_t)i * 8) = o;
  }
}

// ---------------------------------------------------------------------------
// GEMM: C[M,N] = A[M,K] * W[N,K]^T + bias[N]; A,W f16; out f16 or f32.
// Block 256 (8 waves); block tile 128x256, wave tile 64x64.
// Double-buffered LDS, all staging via async global->LDS b128.
// ---------------------------------------------------------------------------
template <bool OUT_F16>
__global__ __launch_bounds__(256) void gemm_bias_f16(
    const _Float16* __restrict__ A, const _Float16* __restrict__ W,
    const float* __restrict__ bias, void* __restrict__ Cout,
    int M, int N, int K) {
  __shared__ alignas(64) _Float16 As[2][128 * 32];
  __shared__ alignas(64) _Float16 Bs[2][256 * 32];

  const int tid  = threadIdx.x;
  const int lane = tid & 31;
  const int wave = tid >> 5;
  const int half = lane >> 4;   // 0: lanes 0-15, 1: lanes 16-31
  const int l16  = lane & 15;
  const int m0   = blockIdx.y * 128;
  const int n0   = blockIdx.x * 256;
  const int wm   = (wave >> 2) * 64;
  const int wn   = (wave & 3) * 64;

  v8f acc[4][4];
#pragma unroll
  for (int i = 0; i < 4; ++i)
#pragma unroll
    for (int j = 0; j < 4; ++j)
#pragma unroll
      for (int r = 0; r < 8; ++r) acc[i][j][r] = 0.0f;

  // staging geometry: A 128x32 -> 16 halfs/thread; W 256x32 -> 32 halfs/thread
  const int srow = tid >> 1;
  const int scol = (tid & 1) * 16;
  int gr = m0 + srow;
  if (gr >= M) gr = 0;  // clamp; stores guarded later
  const unsigned avoff0 = ((unsigned)gr * (unsigned)K + (unsigned)scol) * 2u;
  const unsigned wvoff0 = ((unsigned)(n0 + tid) * (unsigned)K) * 2u;

  auto stage = [&](int kt, int buf) {
    const unsigned koff = (unsigned)kt * 64u;  // 32 halfs
    const unsigned alds = (unsigned)(size_t)&As[buf][srow * 32 + scol];
    const unsigned av   = avoff0 + koff;
    ASYNC_LD(alds, av, (const void*)A, "0");
    ASYNC_LD(alds, av, (const void*)A, "16");
    const unsigned wlds = (unsigned)(size_t)&Bs[buf][tid * 32];
    const unsigned wv   = wvoff0 + koff;
    ASYNC_LD(wlds, wv, (const void*)W, "0");
    ASYNC_LD(wlds, wv, (const void*)W, "16");
    ASYNC_LD(wlds, wv, (const void*)W, "32");
    ASYNC_LD(wlds, wv, (const void*)W, "48");
  };

  auto compute = [&](const _Float16* Asb, const _Float16* Bsb) {
    v16h af[4], bf[4];
#pragma unroll
    for (int i = 0; i < 4; ++i) {
      const _Float16* base = Asb + (wm + i * 16 + l16) * 32;
      af[i] = *(const v16h*)(base + half * 16);
    }
#pragma unroll
    for (int j = 0; j < 4; ++j)
      bf[j] = *(const v16h*)(Bsb + (wn + j * 16 + l16) * 32 + half * 16);
#pragma unroll
    for (int i = 0; i < 4; ++i)
#pragma unroll
      for (int j = 0; j < 4; ++j)
        acc[i][j] = wmma16x16x32(af[i], bf[j], acc[i][j]);
  };

  const int nk = K / 32;
  stage(0, 0);
  wait_async0();
  __syncthreads();
  for (int kt = 0; kt < nk; ++kt) {
    const int cur = kt & 1;
    const bool hasNext = (kt + 1) < nk;
    if (hasNext) stage(kt + 1, cur ^ 1);   // overlap with compute below
    compute(&As[cur][0], &Bs[cur][0]);
    if (hasNext) wait_async0();
    __syncthreads();
  }

  // epilogue (C/D layout: lane = column, VGPR idx = row, half-wave -> +8)
#pragma unroll
  for (int j = 0; j < 4; ++j) {
    const int ncol = n0 + wn + j * 16 + l16;
    const float bv = bias[ncol];
#pragma unroll
    for (int i = 0; i < 4; ++i) {
      const int rbase = m0 + wm + i * 16 + half * 8;
#pragma unroll
      for (int r = 0; r < 8; ++r) {
        const int row = rbase + r;
        if (row < M) {
          const float val = acc[i][j][r] + bv;
          if constexpr (OUT_F16)
            ((_Float16*)Cout)[(size_t)row * N + ncol] = (_Float16)val;
          else
            ((float*)Cout)[(size_t)row * N + ncol] = val;
        }
      }
    }
  }
}

// ---------------------------------------------------------------------------
// Flash-style attention. qkv: f16 [B,T,3,H,64]; out: f16 [B,T,768].
// Grid (ceil(T/128), H, B), block 256 (8 waves); wave owns 16 query rows.
// Double-buffered K (async copy; stale bytes in masked columns are killed by
// the -1e30 mask) and V (register transpose, software-pipelined).
// ---------------------------------------------------------------------------
__global__ __launch_bounds__(256) void attention_kernel(
    const _Float16* __restrict__ qkv, _Float16* __restrict__ out) {
  __shared__ alignas(64) _Float16 Ks[2][64 * 64];   // [key][hd]
  __shared__ alignas(64) _Float16 Vt[2][64 * 64];   // [hd][key]
  __shared__ alignas(64) _Float16 Ps[8][16 * 64];   // per-wave P scratch

  const int tid  = threadIdx.x;
  const int lane = tid & 31;
  const int wave = tid >> 5;
  const int half = lane >> 4;
  const int l16  = lane & 15;
  const int b    = blockIdx.z;
  const int h    = blockIdx.y;
  const int qbase = blockIdx.x * 128 + wave * 16;

  // resident Q fragments (hd chunks 0-31, 32-63)
  int qrow = qbase + l16;
  if (qrow >= SEQ) qrow = SEQ - 1;
  const _Float16* qp = qkv + (size_t)(b * SEQ + qrow) * (3 * DIMX) + h * HDIM;
  v16h qfrag[2];
#pragma unroll
  for (int kk = 0; kk < 2; ++kk) {
    const _Float16* p = qp + kk * 32;
    qfrag[kk] = *(const v16h*)(p + half * 16);
  }

  v8f o[4];
  float mrow[8], lsum[8];
#pragma unroll
  for (int c = 0; c < 4; ++c)
#pragma unroll
    for (int r = 0; r < 8; ++r) o[c][r] = 0.0f;
#pragma unroll
  for (int r = 0; r < 8; ++r) { mrow[r] = -1e30f; lsum[r] = 0.0f; }

  const int srow = tid >> 2;        // 0..63 : key row
  const int scol = (tid & 3) * 16;  // 0..48 : hd segment
  const _Float16* kbase =
      qkv + (size_t)b * SEQ * (3 * DIMX) + DIMX + h * HDIM;

  auto stageK = [&](int kt, int buf) {
    const unsigned klds = (unsigned)(size_t)&Ks[buf][srow * 64 + scol];
    const unsigned kv = (unsigned)(kt * 64 + srow) * (unsigned)(3 * DIMX * 2) +
                        (unsigned)(scol * 2);
    ASYNC_LD(klds, kv, (const void*)kbase, "0");
    ASYNC_LD(klds, kv, (const void*)kbase, "16");
  };
  auto loadV = [&](int kt, v8h& vlo, v8h& vhi) {
    const int grow = kt * 64 + srow;
    if (grow < SEQ) {
      const _Float16* vp = qkv + (size_t)(b * SEQ + grow) * (3 * DIMX) +
                           2 * DIMX + h * HDIM + scol;
      vlo = *(const v8h*)vp;
      vhi = *(const v8h*)(vp + 8);
    } else {
      v8h z;
#pragma unroll
      for (int i = 0; i < 8; ++i) z[i] = (_Float16)0.0f;
      vlo = z; vhi = z;
    }
  };
  auto storeV = [&](int buf, v8h vlo, v8h vhi) {
#pragma unroll
    for (int i = 0; i < 8; ++i) {
      Vt[buf][(scol + i) * 64 + srow]     = vlo[i];
      Vt[buf][(scol + i + 8) * 64 + srow] = vhi[i];
    }
  };

  // prologue: stage tile 0
  stageK(0, 0);
  {
    v8h vlo, vhi;
    loadV(0, vlo, vhi);
    storeV(0, vlo, vhi);
  }
  wait_async0();
  __syncthreads();

  const int nkt = (SEQ + 63) / 64;
  for (int kt = 0; kt < nkt; ++kt) {
    const int cur = kt & 1;
    const bool hasNext = (kt + 1) < nkt;
    v8h nvlo, nvhi;
    if (hasNext) {
      stageK(kt + 1, cur ^ 1);     // async, overlaps compute below
      loadV(kt + 1, nvlo, nvhi);   // global loads in flight during compute
    }

    // ---- S = Q * K^T : preload all fragments, then WMMA chain ----
    v16h kf[4][2];
#pragma unroll
    for (int c = 0; c < 4; ++c)
#pragma unroll
      for (int kk = 0; kk < 2; ++kk)
        kf[c][kk] = *(const v16h*)(&Ks[cur][(c * 16 + l16) * 64 + kk * 32 +
                                           half * 16]);
    v8f s[4];
#pragma unroll
    for (int c = 0; c < 4; ++c)
#pragma unroll
      for (int r = 0; r < 8; ++r) s[c][r] = 0.0f;
#pragma unroll
    for (int c = 0; c < 4; ++c)
#pragma unroll
      for (int kk = 0; kk < 2; ++kk)
        s[c] = wmma16x16x32(qfrag[kk], kf[c][kk], s[c]);

    // scale + key mask (kills stale async data in padded columns)
#pragma unroll
    for (int c = 0; c < 4; ++c) {
      const int kglob = kt * 64 + c * 16 + l16;
      const bool oob = (kglob >= SEQ);
#pragma unroll
      for (int r = 0; r < 8; ++r)
        s[c][r] = oob ? -1e30f : s[c][r] * ATTN_SCALE;
    }

    // online softmax: row lives across 16 lanes of one half-wave
    _Float16* pw = &Ps[wave][0];
#pragma unroll
    for (int r = 0; r < 8; ++r) {
      float mx = fmaxf(fmaxf(s[0][r], s[1][r]), fmaxf(s[2][r], s[3][r]));
#pragma unroll
      for (int d = 8; d >= 1; d >>= 1)
        mx = fmaxf(mx, __shfl_xor(mx, d, 16));
      const float mnew  = fmaxf(mrow[r], mx);
      const float alpha = __expf(mrow[r] - mnew);
      mrow[r] = mnew;
      float rs = 0.0f;
#pragma unroll
      for (int c = 0; c < 4; ++c) {
        const float p = __expf(s[c][r] - mnew);
        s[c][r] = p;
        rs += p;
      }
#pragma unroll
      for (int d = 8; d >= 1; d >>= 1)
        rs += __shfl_xor(rs, d, 16);
      lsum[r] = alpha * lsum[r] + rs;
#pragma unroll
      for (int c = 0; c < 4; ++c) o[c][r] *= alpha;
#pragma unroll
      for (int c = 0; c < 4; ++c)
        pw[(r + half * 8) * 64 + c * 16 + l16] = (_Float16)s[c][r];
    }

    // ---- O += P * V : preload fragments, then WMMA chain ----
    v16h pfrag[2];
#pragma unroll
    for (int kk = 0; kk < 2; ++kk) {
      const _Float16* p = &pw[l16 * 64 + kk * 32];
      pfrag[kk] = *(const v16h*)(p + half * 16);
    }
    v16h vf[4][2];
#pragma unroll
    for (int c = 0; c < 4; ++c)
#pragma unroll
      for (int kk = 0; kk < 2; ++kk)
        vf[c][kk] = *(const v16h*)(&Vt[cur][(c * 16 + l16) * 64 + kk * 32 +
                                           half * 16]);
#pragma unroll
    for (int c = 0; c < 4; ++c)
#pragma unroll
      for (int kk = 0; kk < 2; ++kk)
        o[c] = wmma16x16x32(pfrag[kk], vf[c][kk], o[c]);

    if (hasNext) {
      storeV(cur ^ 1, nvlo, nvhi);
      wait_async0();
    }
    __syncthreads();
  }

  // normalize + store f16 [B,T,768] with head concat
  float inv[8];
#pragma unroll
  for (int r = 0; r < 8; ++r) inv[r] = 1.0f / lsum[r];
#pragma unroll
  for (int c = 0; c < 4; ++c)
#pragma unroll
    for (int r = 0; r < 8; ++r) {
      const int row = qbase + r + half * 8;
      if (row < SEQ)
        out[(size_t)(b * SEQ + row) * DIMX + h * HDIM + c * 16 + l16] =
            (_Float16)(o[c][r] * inv[r]);
    }
}

// ---------------------------------------------------------------------------
extern "C" void kernel_launch(void* const* d_in, const int* in_sizes, int n_in,
                              void* d_out, int out_size, void* d_ws,
                              size_t ws_size, hipStream_t stream) {
  (void)in_sizes; (void)n_in; (void)out_size; (void)ws_size;
  const float* x      = (const float*)d_in[0];
  const float* qkv_w  = (const float*)d_in[1];
  const float* qkv_b  = (const float*)d_in[2];
  const float* proj_w = (const float*)d_in[3];
  const float* proj_b = (const float*)d_in[4];
  float* outp = (float*)d_out;

  const int M = BATCH * SEQ;  // 18464

  // workspace layout (all f16, 256B-aligned)
  size_t off = 0;
  auto walloc = [&](size_t elems) {
    _Float16* p = (_Float16*)((char*)d_ws + off);
    off += ((elems * 2 + 255) & ~(size_t)255);
    return p;
  };
  _Float16* xh     = walloc((size_t)M * DIMX);
  _Float16* qkvwh  = walloc((size_t)3 * DIMX * DIMX);
  _Float16* projwh = walloc((size_t)DIMX * DIMX);
  _Float16* qkvh   = walloc((size_t)M * 3 * DIMX);
  _Float16* attnh  = walloc((size_t)M * DIMX);

  dim3 blk(256);
  const int mblk = (M + 127) / 128;

  // 0) one-shot f32->f16 conversions so all GEMM staging is pure async copy
  {
    const int nx = (M * DIMX) / 8;
    cvt_f32_f16<<<dim3((nx + 255) / 256), blk, 0, stream>>>(x, xh, nx);
    const int nqw = (3 * DIMX * DIMX) / 8;
    cvt_f32_f16<<<dim3((nqw + 255) / 256), blk, 0, stream>>>(qkv_w, qkvwh, nqw);
    const int npw = (DIMX * DIMX) / 8;
    cvt_f32_f16<<<dim3((npw + 255) / 256), blk, 0, stream>>>(proj_w, projwh, npw);
  }

  // 1) QKV projection (N = 2304 -> 9 col-blocks)
  gemm_bias_f16<true><<<dim3((3 * DIMX) / 256, mblk), blk, 0, stream>>>(
      xh, qkvwh, qkv_b, (void*)qkvh, M, 3 * DIMX, DIMX);

  // 2) attention
  attention_kernel<<<dim3((SEQ + 127) / 128, NHEAD, BATCH), blk, 0, stream>>>(
      qkvh, attnh);

  // 3) output projection (N = 768 -> 3 col-blocks), f32 out
  gemm_bias_f16<false><<<dim3(DIMX / 256, mblk), blk, 0, stream>>>(
      attnh, projwh, proj_b, (void*)outp, M, DIMX, DIMX);
}
